// HeterogeneousAttention_5007931867457
// MI455X (gfx1250) — compile-verified, matrix operand order checked by reading, not run on hardware
//
#include <hip/hip_runtime.h>
#include <hip/hip_bf16.h>
#include <math.h>

#define BB   4
#define NN   4096
#define KK   32
#define CC   3
#define DD   64
#define DP   128
#define DCH  69      // C + D + 2
#define KPAD 96      // 69 padded to 3*32 for bf16 WMMA k-steps
#define LRALPHA 0.2f

typedef __attribute__((ext_vector_type(16))) __bf16 v16bf;
typedef __attribute__((ext_vector_type(8)))  __bf16 bf16x8;
typedef __attribute__((ext_vector_type(8)))  float  v8f;

union FragBF { v16bf v; bf16x8 h[2]; __bf16 e[16]; };

// K offset of fragment element-pair j within a 16x16x32 bf16 A/B fragment
__device__ __forceinline__ int koff(int j) { return (j < 4) ? 2*j : 16 + 2*(j - 4); }

// A-matrix fragment from LDS (row-major bf16 tile, stride ld elements).
// lane L<16 holds row L, K = kb+0..7 and kb+16..23 ; lanes 16..31 get kb+8 via caller.
__device__ __forceinline__ v16bf lds_a_frag(const __bf16* base, int ld, int row, int kb) {
  FragBF f;
  const __bf16* p = base + row*ld + kb;
  f.h[0] = *(const bf16x8*)(p);        // K = kb .. kb+7
  f.h[1] = *(const bf16x8*)(p + 16);   // K = kb+16 .. kb+23
  return f.v;
}

// Pre-swizzled B fragment: 16 consecutive bf16 per lane (two ds_load_b128)
__device__ __forceinline__ v16bf lds_frag32(const __bf16* p) {
  FragBF f;
  f.h[0] = *(const bf16x8*)(p);
  f.h[1] = *(const bf16x8*)(p + 8);
  return f.v;
}

// ---------------------------------------------------------------------------
// Kernel 1: disf = concat(pg, pf, dg, df) @ w2 + b2        [B*N blocks]
// ---------------------------------------------------------------------------
__global__ __launch_bounds__(256) void k_disf(
    const float* __restrict__ g,  const float* __restrict__ f,
    const float* __restrict__ gn, const float* __restrict__ fn,
    const float* __restrict__ w2, const float* __restrict__ b2,
    float* __restrict__ disf)
{
  __shared__ __align__(16) float  fS[DD];
  __shared__ __align__(16) float  gS[4];
  __shared__ __align__(16) float  gnS[KK*4];
  __shared__ __align__(16) float  fnS[KK*DD];
  __shared__ __align__(16) __bf16 dgfA[KK*104];        // 32 x 96 (stride 104)
  __shared__ __align__(16) __bf16 w2F[8*3*32*16];      // B fragments, swizzled

  const int tid = threadIdx.x;
  const long long pt = blockIdx.x;                     // b*N + n
  const float* gp  = g  + pt*CC;
  const float* fp  = f  + pt*DD;
  const float* gnp = gn + pt*KK*CC;
  const float* fnp = fn + pt*KK*DD;

  if (tid < DD) fS[tid] = fp[tid];
  if (tid < CC) gS[tid] = gp[tid];
  for (int i = tid; i < KK*CC; i += 256) gnS[(i/CC)*4 + (i%CC)] = gnp[i];
  for (int i = tid; i < KK*DD; i += 256) fnS[i] = fnp[i];

  // Stage w2 as bf16 in WMMA B-fragment order; zero-pad rows 69..95
  for (int idx = tid; idx < 8*3*32*16; idx += 256) {
    int t = idx & 15, lane = (idx >> 4) & 31;
    int ks = (idx >> 9) % 3, ni = idx / (16*32*3);
    int kb = ((lane >> 4) << 3) + ks*32;
    int Krow = kb + koff(t >> 1) + (t & 1);
    int col  = ni*16 + (lane & 15);
    float v = (Krow < DCH) ? w2[Krow*DP + col] : 0.f;
    w2F[idx] = (__bf16)v;
  }
  __syncthreads();

  // Build dgf rows (pg | pf | dg | df | zero-pad) as bf16
  for (int idx = tid; idx < KK*KPAD; idx += 256) {
    int k = idx / KPAD, c = idx % KPAD;
    float v;
    if (c < CC) {
      v = gS[c] - gnS[k*4 + c];
    } else if (c < CC + DD) {
      int d = c - CC; v = fS[d] - fnS[k*DD + d];
    } else if (c == 67) {               // dg = ||pg - gn|| = ||g - 2*gn||
      float s = 0.f;
      #pragma unroll
      for (int q = 0; q < CC; ++q) { float x = gS[q] - 2.f*gnS[k*4+q]; s += x*x; }
      v = sqrtf(s);
    } else if (c == 68) {               // df = ||f - 2*fn||
      float s = 0.f;
      for (int q = 0; q < DD; ++q) { float x = fS[q] - 2.f*fnS[k*DD+q]; s += x*x; }
      v = sqrtf(s);
    } else {
      v = 0.f;                          // K padding 69..95
    }
    dgfA[k*104 + c] = (__bf16)v;
  }
  __syncthreads();

  const int wave = tid >> 5, lane = tid & 31;
  const int ni = wave;
  const int rlo = lane & 15;
  const int kbase = (lane >> 4) << 3;
  v8f acc0 = {}, acc1 = {};
  #pragma unroll
  for (int ks = 0; ks < 3; ++ks) {
    v16bf bfrag = lds_frag32(&w2F[((ni*3 + ks)*32 + lane)*16]);
    v16bf a0 = lds_a_frag(dgfA, 104, rlo,      kbase + ks*32);
    v16bf a1 = lds_a_frag(dgfA, 104, rlo + 16, kbase + ks*32);
    acc0 = __builtin_amdgcn_wmma_f32_16x16x32_bf16(false, a0, false, bfrag, (short)0, acc0, false, false);
    acc1 = __builtin_amdgcn_wmma_f32_16x16x32_bf16(false, a1, false, bfrag, (short)0, acc1, false, false);
  }
  const int col = ni*16 + rlo;
  const float bias = b2[col];
  float* dout = disf + pt*KK*DP;
  #pragma unroll
  for (int v = 0; v < 8; ++v) {
    int m0 = v + kbase;                 // C/D layout: lanes>=16 hold rows +8
    dout[(m0     )*DP + col] = acc0[v] + bias;
    dout[(m0 + 16)*DP + col] = acc1[v] + bias;
  }
}

// ---------------------------------------------------------------------------
// Kernel 2: e = leakyrelu(nfn @ a); att = softmax_k(e); pooled = sum_k att*nfn
//           4 points per block; wave w owns 16-column slab ni=w for all points
// ---------------------------------------------------------------------------
__global__ __launch_bounds__(256) void k_att(
    const float* __restrict__ nfn, const float* __restrict__ amat,
    float* __restrict__ pooled)   // staged into the `out` region [B*N][128]
{
  __shared__ __align__(16) __bf16 nfnS[4*KK*136];      // 34.8 KB

  const int tid = threadIdx.x;
  const long long gp0 = (long long)blockIdx.x * 4;
  const float* src = nfn + gp0*KK*DP;
  for (int idx = tid; idx < 4*KK*DP; idx += 256) {
    int p = idx >> 12, rem = idx & 4095;
    int k = rem >> 7, c = rem & 127;
    nfnS[(p*KK + k)*136 + c] = (__bf16)src[idx];
  }
  __syncthreads();

  const int wave = tid >> 5, lane = tid & 31;
  const int ni = wave, rlo = lane & 15;
  const int kbase = (lane >> 4) << 3;
  const int col = ni*16 + rlo;

  // B fragments of `a` (L2-resident), held in registers across all 4 points
  v16bf bfr[4];
  #pragma unroll
  for (int ks = 0; ks < 4; ++ks) {
    FragBF fr;
    #pragma unroll
    for (int j = 0; j < 8; ++j) {
      int k0 = kbase + ks*32 + koff(j);
      fr.e[2*j]     = (__bf16)amat[(k0    )*DP + col];
      fr.e[2*j + 1] = (__bf16)amat[(k0 + 1)*DP + col];
    }
    bfr[ks] = fr.v;
  }

  for (int p = 0; p < 4; ++p) {
    const __bf16* ab = &nfnS[p*KK*136];
    v8f acc0 = {}, acc1 = {};
    #pragma unroll
    for (int ks = 0; ks < 4; ++ks) {
      v16bf a0 = lds_a_frag(ab, 136, rlo,      kbase + ks*32);
      v16bf a1 = lds_a_frag(ab, 136, rlo + 16, kbase + ks*32);
      acc0 = __builtin_amdgcn_wmma_f32_16x16x32_bf16(false, a0, false, bfr[ks], (short)0, acc0, false, false);
      acc1 = __builtin_amdgcn_wmma_f32_16x16x32_bf16(false, a1, false, bfr[ks], (short)0, acc1, false, false);
    }
    // leaky relu; each lane holds 16 k-rows of one column
    float ev[16];
    #pragma unroll
    for (int t = 0; t < 8; ++t) {
      float x0 = acc0[t]; ev[t]     = (x0 > 0.f) ? x0 : LRALPHA*x0;
      float x1 = acc1[t]; ev[t + 8] = (x1 > 0.f) ? x1 : LRALPHA*x1;
    }
    // softmax over k (32 rows): local 16-reduce + cross-half shfl_xor(16)
    float mx = ev[0];
    #pragma unroll
    for (int t = 1; t < 16; ++t) mx = fmaxf(mx, ev[t]);
    mx = fmaxf(mx, __shfl_xor(mx, 16, 32));
    float s = 0.f;
    #pragma unroll
    for (int t = 0; t < 16; ++t) { ev[t] = __expf(ev[t] - mx); s += ev[t]; }
    s += __shfl_xor(s, 16, 32);
    float inv = 1.f / s;
    // weighted pool against nfn
    float ps = 0.f;
    #pragma unroll
    for (int t = 0; t < 16; ++t) {
      int m = (t < 8) ? (t + kbase) : (t - 8 + kbase + 16);
      ps += ev[t] * inv * (float)nfnS[(p*KK + m)*136 + col];
    }
    ps += __shfl_xor(ps, 16, 32);
    if (lane < 16)
      pooled[(gp0 + p)*DP + col] = ps;
  }
}

// ---------------------------------------------------------------------------
// Kernel 3: out = pooled @ w1 + b1, in-place over the out region.
//           Safe: each block reads only its own 64 rows into LDS pre-barrier.
// ---------------------------------------------------------------------------
__global__ __launch_bounds__(256) void k_out(
    const float* __restrict__ w1, const float* __restrict__ b1,
    float* __restrict__ out)
{
  __shared__ __align__(16) __bf16 At[64*136];          // 17.4 KB
  __shared__ __align__(16) __bf16 w1F[8*4*32*16];      // 32 KB, swizzled B frags

  const int tid = threadIdx.x;
  const long long row0 = (long long)blockIdx.x * 64;

  for (int idx = tid; idx < 64*DP; idx += 256) {
    int r = idx >> 7, c = idx & 127;
    At[r*136 + c] = (__bf16)out[(row0 + r)*DP + c];
  }
  for (int idx = tid; idx < 8*4*32*16; idx += 256) {
    int t = idx & 15, lane = (idx >> 4) & 31;
    int ks = (idx >> 9) & 3, ni = idx >> 11;
    int kb = ((lane >> 4) << 3) + ks*32;
    int Krow = kb + koff(t >> 1) + (t & 1);
    int col  = ni*16 + (lane & 15);
    w1F[idx] = (__bf16)w1[Krow*DP + col];
  }
  __syncthreads();

  const int wave = tid >> 5, lane = tid & 31;
  const int ni = wave, rlo = lane & 15;
  const int kbase = (lane >> 4) << 3;
  v8f acc[4] = {{}, {}, {}, {}};
  #pragma unroll
  for (int ks = 0; ks < 4; ++ks) {
    v16bf bfrag = lds_frag32(&w1F[((ni*4 + ks)*32 + lane)*16]);
    #pragma unroll
    for (int mi = 0; mi < 4; ++mi) {
      v16bf af = lds_a_frag(At, 136, mi*16 + rlo, kbase + ks*32);
      acc[mi] = __builtin_amdgcn_wmma_f32_16x16x32_bf16(false, af, false, bfrag, (short)0, acc[mi], false, false);
    }
  }
  const int col = ni*16 + rlo;
  const float bias = b1[col];
  #pragma unroll
  for (int mi = 0; mi < 4; ++mi) {
    #pragma unroll
    for (int v = 0; v < 8; ++v) {
      int m = mi*16 + v + kbase;
      out[(row0 + m)*DP + col] = acc[mi][v] + bias;
    }
  }
}

// ---------------------------------------------------------------------------
extern "C" void kernel_launch(void* const* d_in, const int* in_sizes, int n_in,
                              void* d_out, int out_size, void* d_ws, size_t ws_size,
                              hipStream_t stream) {
  (void)in_sizes; (void)n_in; (void)out_size; (void)d_ws; (void)ws_size;
  const float* g   = (const float*)d_in[0];
  const float* f   = (const float*)d_in[1];
  const float* gn  = (const float*)d_in[2];
  const float* fn  = (const float*)d_in[3];
  const float* nfn = (const float*)d_in[4];
  const float* a   = (const float*)d_in[5];
  const float* w2  = (const float*)d_in[6];
  const float* b2  = (const float*)d_in[7];
  const float* w1  = (const float*)d_in[8];
  const float* b1  = (const float*)d_in[9];

  float* out  = (float*)d_out;                       // [B*N][128]
  float* disf = out + (long long)BB*NN*DP;           // [B*N*K][128]

  k_disf<<<BB*NN,      256, 0, stream>>>(g, f, gn, fn, w2, b2, disf);
  k_att <<<BB*NN/4,    256, 0, stream>>>(nfn, a, out);   // writes pooled into out
  k_out <<<BB*NN/64,   256, 0, stream>>>(w1, b1, out);   // in-place pooled@w1+b1
}
